// CrystalGraphConvNet_34282428956968
// MI455X (gfx1250) — matrix operand order repaired and checked by reading, not compiled, not run on hardware
//
#include <hip/hip_runtime.h>
#include <hip/hip_bf16.h>

typedef _Float16 v16h __attribute__((ext_vector_type(16)));
typedef _Float16 v8h  __attribute__((ext_vector_type(8)));
typedef float    v8f  __attribute__((ext_vector_type(8)));

#define FN 64      // node feature len
#define FE 64      // edge feature len
#define MM 24      // neighbors
#define KTOT 192   // 2F + E
#define NOUT 128   // 2F

__device__ __forceinline__ float softplusf(float x) {
    return fmaxf(x, 0.0f) + log1pf(expf(-fabsf(x)));
}
__device__ __forceinline__ float sigmoidf(float x) {
    return 1.0f / (1.0f + expf(-x));
}

// node = node_fea @ Wn + bn   (N x 92) @ (92 x 64)
__global__ __launch_bounds__(256) void embed_node_k(
    const float* __restrict__ nf, const float* __restrict__ Wn,
    const float* __restrict__ bn, float* __restrict__ out, int n) {
    int t = blockIdx.x * 256 + threadIdx.x;
    if (t >= n * FN) return;
    int i = t >> 6, c = t & 63;
    float s = bn[c];
    const float* row = nf + (size_t)i * 92;
    #pragma unroll 4
    for (int k = 0; k < 92; ++k) s += row[k] * Wn[k * FN + c];
    out[t] = s;
}

// edge = f16(edge_fea @ We + be)   (N*M x 41) @ (41 x 64)
__global__ __launch_bounds__(256) void embed_edge_k(
    const float* __restrict__ ef, const float* __restrict__ We,
    const float* __restrict__ be, _Float16* __restrict__ out, int rows) {
    int t = blockIdx.x * 256 + threadIdx.x;
    if (t >= rows * FE) return;
    int r = t >> 6, c = t & 63;
    float s = be[c];
    const float* row = ef + (size_t)r * 41;
    #pragma unroll
    for (int k = 0; k < 41; ++k) s += row[k] * We[k * FE + c];
    out[t] = (_Float16)s;
}

// Pack W (192x128 f32, row-major) into WMMA B-fragment order (f16):
// dst[((kb*8+ct)*32+lane)*8 + j] = pack(W[K0][c], W[K0+1][c])
// with c = ct*16 + (lane&15), K0 = kb*32 + (lane<16 ? 0 : 16) + 2j
// (contiguous k order, matched by A: lanes 0-15 hold K 0..15, lanes 16-31 hold K 16..31)
__global__ __launch_bounds__(256) void pack_w_k(
    const float* __restrict__ W, unsigned int* __restrict__ dst) {
    int t = blockIdx.x * 256 + threadIdx.x;
    if (t >= 6 * 8 * 32 * 8) return;
    int j    = t & 7;
    int lane = (t >> 3) & 31;
    int ct   = (t >> 8) & 7;
    int kb   = t >> 11;
    int c  = ct * 16 + (lane & 15);
    int K0 = kb * 32 + ((lane < 16) ? 0 : 16) + 2 * j;
    union { _Float16 h[2]; unsigned int u; } p;
    p.h[0] = (_Float16)W[K0 * NOUT + c];
    p.h[1] = (_Float16)W[(K0 + 1) * NOUT + c];
    dst[t] = p.u;
}

// One workgroup per node: 32(padded M) x 192 A tile in LDS,
// 2 row-tiles x 8 col-tiles of 16x16 WMMA f32<-f16 outputs,
// then gated activation + neighbor reduction + softplus.
// Edge block (already f16) is copied global->LDS with the CDNA5
// async-LDS path (GLOBAL_LOAD_ASYNC_TO_LDS_B128, ASYNCcnt-tracked).
__global__ __launch_bounds__(256) void conv_k(
    const float* __restrict__ nodeIn, const _Float16* __restrict__ edgeh,
    const int* __restrict__ eidx, const unsigned int* __restrict__ packedW,
    const float* __restrict__ bias, const float* __restrict__ alphaPtr,
    float* __restrict__ nodeOut) {
    __shared__ __align__(16) _Float16 As[32 * 200]; // padded stride vs bank conflicts
    __shared__ float Cs[32 * NOUT];
    __shared__ int   idxs[32];
    __shared__ float msk[32];
    __shared__ float part[4 * FN];

    const int i    = blockIdx.x;
    const int tid  = threadIdx.x;
    const int lane = tid & 31;
    const int ct   = tid >> 5;   // wave id = output column tile

    // Resident B fragments for this wave's column tile (6 K-chunks).
    v16h bfrag[6];
    const v16h* wp = reinterpret_cast<const v16h*>(packedW);
    #pragma unroll
    for (int kb = 0; kb < 6; ++kb) bfrag[kb] = wp[(kb * 8 + ct) * 32 + lane];

    // Async copy of the edge block: A columns [128..191] for all 32 rows.
    // thread t -> row r = t/8, 16-byte segment seg = t%8.
    // Rows 24..31 read adjacent (finite) memory; masked out after the GEMM.
    {
        const int r   = tid >> 3;
        const int seg = tid & 7;
        unsigned int ldsa = (unsigned int)(size_t)(&As[r * 200 + 2 * FN + seg * 8]);
        unsigned long long ga =
            (unsigned long long)(edgeh + ((size_t)i * MM + r) * FE + seg * 8);
        asm volatile("global_load_async_to_lds_b128 %0, %1, off"
                     :: "v"(ldsa), "v"(ga) : "memory");
    }

    if (tid < 32) {
        int v = (tid < MM) ? eidx[i * MM + tid] : -1;
        idxs[tid] = v;
        msk[tid]  = (v >= 0) ? 1.0f : 0.0f;
    }
    __syncthreads();

    // Build self|nbr part of A (cols 0..127) as f16 pairs; overlaps async copy.
    for (int e2 = tid; e2 < 32 * FN; e2 += 256) {   // 64 pairs per row
        int r  = e2 >> 6;
        int cc = (e2 & 63) * 2;
        float2 v;
        if (cc < FN) {
            v = *reinterpret_cast<const float2*>(nodeIn + (size_t)i * FN + cc);
        } else {
            int jn = idxs[r]; jn = (jn < 0) ? 0 : jn;
            v = *reinterpret_cast<const float2*>(nodeIn + (size_t)jn * FN + (cc - FN));
        }
        union { _Float16 h[2]; unsigned int u; } p;
        p.h[0] = (_Float16)v.x; p.h[1] = (_Float16)v.y;
        *reinterpret_cast<unsigned int*>(&As[r * 200 + cc]) = p.u;
    }

    // All async LDS writes visible before the barrier releases readers.
    asm volatile("s_wait_asynccnt 0x0" ::: "memory");
    __syncthreads();

    // GEMM: two 16-row tiles, K = 6 x 32, uniform control flow (EXEC all ones).
    #pragma unroll
    for (int rt = 0; rt < 2; ++rt) {
        v8f acc = {0.f, 0.f, 0.f, 0.f, 0.f, 0.f, 0.f, 0.f};
        const int rowbase = (rt * 16 + (lane & 15)) * 200;
        const int koff    = (lane < 16) ? 0 : 16;
        #pragma unroll
        for (int kb = 0; kb < 6; ++kb) {
            // A layout matching packed B: lane<16 K{0..15}, lane>=16 K{16..31}
            v8h lo = *reinterpret_cast<const v8h*>(&As[rowbase + kb * 32 + koff]);
            v8h hi = *reinterpret_cast<const v8h*>(&As[rowbase + kb * 32 + 8 + koff]);
            v16h afrag = __builtin_shufflevector(lo, hi,
                0, 1, 2, 3, 4, 5, 6, 7, 8, 9, 10, 11, 12, 13, 14, 15);
            acc = __builtin_amdgcn_wmma_f32_16x16x32_f16(
                false, afrag, false, bfrag[kb], (short)0, acc, false, false);
        }
        // D layout: VGPR k -> row k + 8*(lane>=16), col = lane&15
        const int r0    = rt * 16 + ((lane < 16) ? 0 : 8);
        const int cbase = ct * 16 + (lane & 15);
        #pragma unroll
        for (int k = 0; k < 8; ++k) Cs[(r0 + k) * NOUT + cbase] = acc[k];
    }
    __syncthreads();

    // prod = sigmoid(filter) * softplus(core) * mask ; sum over neighbors.
    {
        const int c  = tid & 63;
        const int rg = tid >> 6;
        float s = 0.f;
        #pragma unroll
        for (int r = rg * 8; r < rg * 8 + 8; ++r) {
            float g1 = Cs[r * NOUT + c] + bias[c];
            float g2 = Cs[r * NOUT + FN + c] + bias[FN + c];
            s += sigmoidf(g1) * softplusf(g2) * msk[r];
        }
        part[rg * FN + c] = s;
    }
    __syncthreads();
    if (tid < FN) {
        float s = part[tid] + part[FN + tid] + part[2 * FN + tid] + part[3 * FN + tid];
        float x = alphaPtr[0] * nodeIn[(size_t)i * FN + tid] + s;
        nodeOut[(size_t)i * FN + tid] = softplusf(x);
    }
}

extern "C" void kernel_launch(void* const* d_in, const int* in_sizes, int n_in,
                              void* d_out, int out_size, void* d_ws, size_t ws_size,
                              hipStream_t stream) {
    const float* node_fea = (const float*)d_in[0];
    const float* edge_fea = (const float*)d_in[1];
    const int*   eidx     = (const int*)d_in[2];
    const float* Wn = (const float*)d_in[3];
    const float* bn = (const float*)d_in[4];
    const float* We = (const float*)d_in[5];
    const float* be = (const float*)d_in[6];
    const float* W1 = (const float*)d_in[7];
    const float* b1 = (const float*)d_in[8];
    const float* a1 = (const float*)d_in[9];
    const float* W2 = (const float*)d_in[10];
    const float* b2 = (const float*)d_in[11];
    const float* a2 = (const float*)d_in[12];
    const float* W3 = (const float*)d_in[13];
    const float* b3 = (const float*)d_in[14];
    const float* a3 = (const float*)d_in[15];

    const int N = in_sizes[0] / 92;

    // workspace layout
    char* ws = (char*)d_ws;
    _Float16* edgeh = (_Float16*)ws;
    size_t off = (size_t)N * MM * FE * sizeof(_Float16);
    off = (off + 255) & ~(size_t)255;
    float* nodeA = (float*)(ws + off);  off += (size_t)N * FN * sizeof(float);
    off = (off + 255) & ~(size_t)255;
    float* nodeB = (float*)(ws + off);  off += (size_t)N * FN * sizeof(float);
    off = (off + 255) & ~(size_t)255;
    unsigned int* pw1 = (unsigned int*)(ws + off); off += 6 * 8 * 32 * 8 * 4;
    unsigned int* pw2 = (unsigned int*)(ws + off); off += 6 * 8 * 32 * 8 * 4;
    unsigned int* pw3 = (unsigned int*)(ws + off);

    embed_node_k<<<(N * FN + 255) / 256, 256, 0, stream>>>(node_fea, Wn, bn, nodeA, N);
    embed_edge_k<<<(N * MM * FE + 255) / 256, 256, 0, stream>>>(edge_fea, We, be, edgeh, N * MM);
    pack_w_k<<<48, 256, 0, stream>>>(W1, pw1);
    pack_w_k<<<48, 256, 0, stream>>>(W2, pw2);
    pack_w_k<<<48, 256, 0, stream>>>(W3, pw3);

    conv_k<<<N, 256, 0, stream>>>(nodeA, edgeh, eidx, pw1, b1, a1, nodeB);
    conv_k<<<N, 256, 0, stream>>>(nodeB, edgeh, eidx, pw2, b2, a2, nodeA);
    conv_k<<<N, 256, 0, stream>>>(nodeA, edgeh, eidx, pw3, b3, a3, (float*)d_out);
}
